// RelationalGraphLayer_42262478192900
// MI455X (gfx1250) — matrix operand order repaired and checked by reading, not compiled, not run on hardware
//
#include <hip/hip_runtime.h>

// ---------------------------------------------------------------------------
// Types for CDNA5 WMMA (wave32)
// ---------------------------------------------------------------------------
typedef __attribute__((ext_vector_type(16))) __bf16 v16bf;
typedef __attribute__((ext_vector_type(8)))  __bf16 v8bf;
typedef __attribute__((ext_vector_type(4)))  __bf16 v4bf;
typedef __attribute__((ext_vector_type(8)))  float  v8f;

#define DHEAD 128
#define NHEAD 4
#define QKDIM 512   // H * D

__device__ __forceinline__ __bf16 f2bf(float f) {
  unsigned u = __float_as_uint(f);
  unsigned r = (u + 0x7FFFu + ((u >> 16) & 1u)) >> 16;   // round-to-nearest-even
  unsigned short h = (unsigned short)r;
  return __builtin_bit_cast(__bf16, h);
}

// monotonic float<->uint encoding so atomicMax(uint) == float max
__device__ __forceinline__ unsigned encF(float f) {
  unsigned u = __float_as_uint(f);
  return (u & 0x80000000u) ? ~u : (u | 0x80000000u);
}
__device__ __forceinline__ float decF(unsigned u) {
  unsigned v = (u & 0x80000000u) ? (u & 0x7FFFFFFFu) : ~u;
  return __uint_as_float(v);
}

// ---------------------------------------------------------------------------
// Elementwise conversion / init kernels
// ---------------------------------------------------------------------------
__global__ void k_f32_to_bf16(const float* __restrict__ in, __bf16* __restrict__ out, long n) {
  long i = (long)blockIdx.x * blockDim.x + threadIdx.x;
  if (i < n) out[i] = f2bf(in[i]);
}

// in: [R, K, M] f32 row-major  ->  out: [R, M, K] bf16 (transposed per relation)
__global__ void k_conv_w_T(const float* __restrict__ in, __bf16* __restrict__ out,
                           int K, int M, int R) {
  long i = (long)blockIdx.x * blockDim.x + threadIdx.x;
  long total = (long)R * K * M;
  if (i >= total) return;
  long km = (long)K * M;
  int  r  = (int)(i / km);
  long rm = i % km;
  int  k  = (int)(rm / M);
  int  n  = (int)(rm % M);
  out[((long)r * M + n) * K + k] = f2bf(in[i]);
}

__global__ void k_zero_f32(float* __restrict__ p, long n) {
  long i = (long)blockIdx.x * blockDim.x + threadIdx.x;
  if (i < n) p[i] = 0.0f;
}

__global__ void k_init_softmax(unsigned* __restrict__ m, float* __restrict__ den, long n) {
  long i = (long)blockIdx.x * blockDim.x + threadIdx.x;
  if (i < n) { m[i] = 0u; den[i] = 0.0f; }
}

// ---------------------------------------------------------------------------
// WMMA GEMM: Y[N,M] = X[N,128](bf16) @ Wt^T + bias   (Wt stored as [M,128] bf16)
// One wave computes a 16x64 strip: 4 accumulators share one A fragment
// (4x A reuse), K=128 in four K=32 WMMA steps -> 16 v_wmma per wave.
// OutT = __bf16 for q/k/v (halves HBM write + later gather traffic),
// OutT = float with ACCUM for the skip projection into the node accumulator.
// ---------------------------------------------------------------------------
template <bool ACCUM, typename OutT>
__global__ void gemm_bf16_wmma(const __bf16* __restrict__ X,   // [N,128]
                               const __bf16* __restrict__ Wt,  // [M,128] (pre-transposed)
                               const float*  __restrict__ bias,// [M]
                               OutT* __restrict__ Y,           // [N,M]
                               int N, int M)
{
  const int wave = (blockIdx.x * blockDim.x + threadIdx.x) >> 5;
  const int lane = threadIdx.x & 31;
  const int half = lane >> 4;     // lane group 0: lanes 0-15, group 1: lanes 16-31
  const int r    = lane & 15;
  const int groups = M >> 6;      // number of 64-wide column groups
  const int tm = (wave / groups) << 4;   // row base
  const int tg = (wave % groups) << 6;   // column base (64-wide strip)
  if (tm >= N) return;

  const __bf16* xrow = X + (size_t)(tm + r) * 128;  // A row M=r

  v8f acc[4];
#pragma unroll
  for (int t = 0; t < 4; ++t) {
    const float bn = bias[tg + t * 16 + r];         // bias by output column
#pragma unroll
    for (int i = 0; i < 8; ++i) acc[t][i] = bn;
  }

#pragma unroll
  for (int kk = 0; kk < 128; kk += 32) {
    // A 16x32 bf16 layout: half 0 -> K {0..7,16..23}, half 1 -> K {8..15,24..31}
    v8bf alo = *(const v8bf*)(xrow + kk + (half << 3));
    v8bf ahi = *(const v8bf*)(xrow + kk + 16 + (half << 3));
    v16bf a;
#pragma unroll
    for (int e = 0; e < 8; ++e) { a[e] = alo[e]; a[8 + e] = ahi[e]; }
#pragma unroll
    for (int t = 0; t < 4; ++t) {
      // B 32x16 bf16 layout: half 0 -> K {0..7,16..23}, half 1 -> K {8..15,24..31}
      const __bf16* wcol = Wt + (size_t)(tg + t * 16 + r) * 128;
      v8bf blo = *(const v8bf*)(wcol + kk + (half << 3));
      v8bf bhi = *(const v8bf*)(wcol + kk + 16 + (half << 3));
      v16bf b;
#pragma unroll
      for (int e = 0; e < 8; ++e) { b[e] = blo[e]; b[8 + e] = bhi[e]; }
      acc[t] = __builtin_amdgcn_wmma_f32_16x16x32_bf16(
          /*neg_a=*/false, a, /*neg_b=*/false, b,
          /*c_mod=*/(short)0, acc[t], /*reuse_a=*/false, /*reuse_b=*/false);
    }
  }

  // C/D layout: lane -> N=r, VGPR i -> row M = i + 8*half
#pragma unroll
  for (int t = 0; t < 4; ++t) {
    OutT* yp = Y + (size_t)tm * M + tg + t * 16 + r;
#pragma unroll
    for (int i = 0; i < 8; ++i) {
      size_t o = (size_t)(i + (half << 3)) * M;
      if constexpr (ACCUM) {
        yp[o] += acc[t][i];
      } else if constexpr (__is_same(OutT, __bf16)) {
        yp[o] = f2bf(acc[t][i]);
      } else {
        yp[o] = acc[t][i];
      }
    }
  }
}

// ---------------------------------------------------------------------------
// Edge pass 1: score[e,h] = dot(q[dst,h,:], k[src,h,:]) / sqrt(D); segment max
// One wave per edge; each lane covers 4 contiguous columns (one 8B v4bf load).
// ---------------------------------------------------------------------------
__global__ void k_edge_score_max(const int* __restrict__ ei, int E,
                                 const __bf16* __restrict__ q,  // [Nd,512]
                                 const __bf16* __restrict__ k,  // [Ns,512]
                                 float* __restrict__ score,     // [E,4]
                                 unsigned* __restrict__ mEnc)   // [Nd,4]
{
  const int wid  = (blockIdx.x * blockDim.x + threadIdx.x) >> 5;
  const int lane = threadIdx.x & 31;
  if (wid >= E) return;
  const int src = ei[wid];
  const int dst = ei[E + wid];
  const float inv_sqrt_d = 0.088388347648318447f;   // 1/sqrt(128)
#pragma unroll
  for (int h = 0; h < NHEAD; ++h) {
    const __bf16* qp = q + (size_t)dst * QKDIM + h * DHEAD + lane * 4;
    const __bf16* kp = k + (size_t)src * QKDIM + h * DHEAD + lane * 4;
    v4bf qv = *(const v4bf*)qp;
    v4bf kv = *(const v4bf*)kp;
    float s = 0.0f;
#pragma unroll
    for (int i = 0; i < 4; ++i) s += (float)qv[i] * (float)kv[i];
    for (int off = 16; off > 0; off >>= 1) s += __shfl_down(s, off, 32);
    if (lane == 0) {
      s *= inv_sqrt_d;
      score[(size_t)wid * NHEAD + h] = s;
      atomicMax(&mEnc[(size_t)dst * NHEAD + h], encF(s));
    }
  }
}

// ---------------------------------------------------------------------------
// Edge pass 2: ex = exp(score - m[dst]); den[dst] += ex  (one thread per (e,h))
// ---------------------------------------------------------------------------
__global__ void k_edge_exp_sum(const int* __restrict__ ei, int E,
                               float* __restrict__ score,         // [E,4] in: score, out: exp
                               const unsigned* __restrict__ mEnc, // [Nd,4]
                               float* __restrict__ den)           // [Nd,4]
{
  long t = (long)blockIdx.x * blockDim.x + threadIdx.x;
  if (t >= (long)E * NHEAD) return;
  int e = (int)(t >> 2), h = (int)(t & 3);
  int dst = ei[E + e];
  float m  = decF(mEnc[(size_t)dst * NHEAD + h]);
  float ex = __expf(score[t] - m);
  score[t] = ex;
  atomicAdd(&den[(size_t)dst * NHEAD + h], ex);
}

// ---------------------------------------------------------------------------
// Edge pass 3: accum[dst,:] += (alpha/H) * v[src,h,:]  (one wave per (e,h))
// ---------------------------------------------------------------------------
__global__ void k_edge_aggregate(const int* __restrict__ ei, int E,
                                 const float* __restrict__ score, // [E,4] = ex
                                 const float* __restrict__ den,   // [Nd,4]
                                 const __bf16* __restrict__ v,    // [Ns,512]
                                 float* __restrict__ accum)       // [Nd,128]
{
  const long gid  = (long)blockIdx.x * blockDim.x + threadIdx.x;
  const long wid  = gid >> 5;
  const int  lane = threadIdx.x & 31;
  if (wid >= (long)E * NHEAD) return;
  const int e = (int)(wid >> 2), h = (int)(wid & 3);
  const int src = ei[e];
  const int dst = ei[E + e];
  const float alpha = score[wid] / den[(size_t)dst * NHEAD + h] * (1.0f / NHEAD);
  const __bf16* vp = v + (size_t)src * QKDIM + h * DHEAD + lane * 4;
  v4bf vv = *(const v4bf*)vp;
  float* op = accum + (size_t)dst * DHEAD + lane * 4;
#pragma unroll
  for (int i = 0; i < 4; ++i) {
    atomicAdd(&op[i], alpha * (float)vv[i]);
  }
}

// ---------------------------------------------------------------------------
// out = LayerNorm(acc) * w + b + x      (one wave per row of 128)
// ---------------------------------------------------------------------------
__global__ void k_ln_residual(const float* __restrict__ acc,  // [N,128]
                              const float* __restrict__ x,    // [N,128]
                              const float* __restrict__ w,    // [128]
                              const float* __restrict__ b,    // [128]
                              float* __restrict__ out, int N)
{
  const int wid  = (blockIdx.x * blockDim.x + threadIdx.x) >> 5;
  const int lane = threadIdx.x & 31;
  if (wid >= N) return;
  const float* a = acc + (size_t)wid * DHEAD;
  float vloc[4];
  float sum = 0.0f;
#pragma unroll
  for (int i = 0; i < 4; ++i) { vloc[i] = a[lane * 4 + i]; sum += vloc[i]; }
  for (int off = 16; off > 0; off >>= 1) sum += __shfl_xor(sum, off, 32);
  const float mu = sum * (1.0f / DHEAD);
  float var = 0.0f;
#pragma unroll
  for (int i = 0; i < 4; ++i) { float d = vloc[i] - mu; var += d * d; }
  for (int off = 16; off > 0; off >>= 1) var += __shfl_xor(var, off, 32);
  var *= (1.0f / DHEAD);
  const float rs = rsqrtf(var + 1e-5f);
#pragma unroll
  for (int i = 0; i < 4; ++i) {
    int c = lane * 4 + i;
    out[(size_t)wid * DHEAD + c] = (vloc[i] - mu) * rs * w[c] + b[c]
                                 + x[(size_t)wid * DHEAD + c];
  }
}

// ---------------------------------------------------------------------------
// Host-side orchestration
// ---------------------------------------------------------------------------
extern "C" void kernel_launch(void* const* d_in, const int* in_sizes, int n_in,
                              void* d_out, int out_size, void* d_ws, size_t ws_size,
                              hipStream_t stream)
{
  const float* x_user  = (const float*)d_in[0];
  const float* x_tweet = (const float*)d_in[1];
  const int* ei_follow = (const int*)d_in[2];
  const int* ei_post   = (const int*)d_in[3];
  const int* ei_rev    = (const int*)d_in[4];
  const float* Wq = (const float*)d_in[5];
  const float* bq = (const float*)d_in[6];
  const float* Wk = (const float*)d_in[7];
  const float* bk = (const float*)d_in[8];
  const float* Wv = (const float*)d_in[9];
  const float* bv = (const float*)d_in[10];
  const float* Ws = (const float*)d_in[11];
  const float* bs = (const float*)d_in[12];
  const float* lnwu = (const float*)d_in[13];
  const float* lnbu = (const float*)d_in[14];
  const float* lnwt = (const float*)d_in[15];
  const float* lnbt = (const float*)d_in[16];

  const int NU = in_sizes[0] / DHEAD;    // 100000
  const int NT = in_sizes[1] / DHEAD;    // 200000
  const int E  = in_sizes[2] / 2;        // 320000

  // ---- carve workspace -----------------------------------------------------
  size_t off = 0;
  auto carve = [&](size_t bytes) -> void* {
    void* p = (char*)d_ws + off;
    off += (bytes + 255) & ~(size_t)255;
    return p;
  };
  __bf16* xb_user  = (__bf16*)carve((size_t)NU * DHEAD * sizeof(__bf16));
  __bf16* xb_tweet = (__bf16*)carve((size_t)NT * DHEAD * sizeof(__bf16));
  __bf16* Wqb = (__bf16*)carve((size_t)3 * QKDIM * DHEAD * sizeof(__bf16)); // [3,512,128] transposed
  __bf16* Wkb = (__bf16*)carve((size_t)3 * QKDIM * DHEAD * sizeof(__bf16));
  __bf16* Wvb = (__bf16*)carve((size_t)3 * QKDIM * DHEAD * sizeof(__bf16));
  __bf16* Wsb = (__bf16*)carve((size_t)3 * DHEAD * DHEAD * sizeof(__bf16)); // [3,128,128] transposed
  __bf16* q_buf = (__bf16*)carve((size_t)NT * QKDIM * sizeof(__bf16));
  __bf16* k_buf = (__bf16*)carve((size_t)NT * QKDIM * sizeof(__bf16));
  __bf16* v_buf = (__bf16*)carve((size_t)NT * QKDIM * sizeof(__bf16));
  float* score = (float*)carve((size_t)E * NHEAD * sizeof(float));
  unsigned* m_buf = (unsigned*)carve((size_t)NT * NHEAD * sizeof(unsigned));
  float* den_buf  = (float*)carve((size_t)NT * NHEAD * sizeof(float));
  float* u_buf = (float*)carve((size_t)NU * DHEAD * sizeof(float));
  float* t_buf = (float*)carve((size_t)NT * DHEAD * sizeof(float));

  const int BLK = 256;
  auto nblk = [&](long threads) { return (unsigned)((threads + BLK - 1) / BLK); };

  // ---- precision conversion ------------------------------------------------
  k_f32_to_bf16<<<nblk((long)NU * DHEAD), BLK, 0, stream>>>(x_user,  xb_user,  (long)NU * DHEAD);
  k_f32_to_bf16<<<nblk((long)NT * DHEAD), BLK, 0, stream>>>(x_tweet, xb_tweet, (long)NT * DHEAD);
  k_conv_w_T<<<nblk(3L * DHEAD * QKDIM), BLK, 0, stream>>>(Wq, Wqb, DHEAD, QKDIM, 3);
  k_conv_w_T<<<nblk(3L * DHEAD * QKDIM), BLK, 0, stream>>>(Wk, Wkb, DHEAD, QKDIM, 3);
  k_conv_w_T<<<nblk(3L * DHEAD * QKDIM), BLK, 0, stream>>>(Wv, Wvb, DHEAD, QKDIM, 3);
  k_conv_w_T<<<nblk(3L * DHEAD * DHEAD), BLK, 0, stream>>>(Ws, Wsb, DHEAD, DHEAD, 3);

  // ---- zero accumulators ---------------------------------------------------
  k_zero_f32<<<nblk((long)NU * DHEAD), BLK, 0, stream>>>(u_buf, (long)NU * DHEAD);
  k_zero_f32<<<nblk((long)NT * DHEAD), BLK, 0, stream>>>(t_buf, (long)NT * DHEAD);

  // qkv projections: bf16 output, no accumulate
  auto gemm_qkv = [&](const __bf16* X, const __bf16* Wt, const float* bias,
                      __bf16* Y, int N, int M) {
    long waves = (long)(N >> 4) * (M >> 6);
    gemm_bf16_wmma<false, __bf16><<<nblk(waves * 32), BLK, 0, stream>>>(X, Wt, bias, Y, N, M);
  };
  // skip projection: f32 accumulate into node buffer
  auto gemm_skip = [&](const __bf16* X, const __bf16* Wt, const float* bias,
                       float* Y, int N, int M) {
    long waves = (long)(N >> 4) * (M >> 6);
    gemm_bf16_wmma<true, float><<<nblk(waves * 32), BLK, 0, stream>>>(X, Wt, bias, Y, N, M);
  };

  // relation r: src features / dst features / edges / accumulator / weight idx
  struct Rel { const __bf16* xs; int Ns; const __bf16* xd; int Nd;
               const int* ei; float* accum; int w; };
  Rel rels[3] = {
    { xb_user,  NU, xb_user,  NU, ei_follow, u_buf, 0 },  // user-follow-user
    { xb_user,  NU, xb_tweet, NT, ei_post,   t_buf, 1 },  // user-post-tweet
    { xb_tweet, NT, xb_user,  NU, ei_rev,    u_buf, 2 },  // tweet-rev-user
  };

  for (int r = 0; r < 3; ++r) {
    const Rel& R = rels[r];
    const size_t wqk_off = (size_t)R.w * QKDIM * DHEAD;
    const size_t ws_off  = (size_t)R.w * DHEAD * DHEAD;

    gemm_qkv(R.xd, Wqb + wqk_off, bq + (size_t)R.w * QKDIM, q_buf, R.Nd, QKDIM);
    gemm_qkv(R.xs, Wkb + wqk_off, bk + (size_t)R.w * QKDIM, k_buf, R.Ns, QKDIM);
    gemm_qkv(R.xs, Wvb + wqk_off, bv + (size_t)R.w * QKDIM, v_buf, R.Ns, QKDIM);
    gemm_skip(R.xd, Wsb + ws_off, bs + (size_t)R.w * DHEAD, R.accum, R.Nd, DHEAD);

    // scatter softmax + aggregation
    k_init_softmax<<<nblk((long)R.Nd * NHEAD), BLK, 0, stream>>>(m_buf, den_buf, (long)R.Nd * NHEAD);
    k_edge_score_max<<<nblk((long)E * 32), BLK, 0, stream>>>(R.ei, E, q_buf, k_buf, score, m_buf);
    k_edge_exp_sum<<<nblk((long)E * NHEAD), BLK, 0, stream>>>(R.ei, E, score, m_buf, den_buf);
    k_edge_aggregate<<<nblk((long)E * NHEAD * 32), BLK, 0, stream>>>(R.ei, E, score, den_buf, v_buf, R.accum);
  }

  // ---- LayerNorm + residual into d_out (out_user then out_tweet) -----------
  float* out_user  = (float*)d_out;
  float* out_tweet = (float*)d_out + (size_t)NU * DHEAD;
  k_ln_residual<<<nblk((long)NU * 32), BLK, 0, stream>>>(u_buf, x_user,  lnwu, lnbu, out_user,  NU);
  k_ln_residual<<<nblk((long)NT * 32), BLK, 0, stream>>>(t_buf, x_tweet, lnwt, lnbt, out_tweet, NT);
}
